// CoAttention_3573412790676
// MI455X (gfx1250) — matrix operand order repaired and checked by reading, not compiled, not run on hardware
//
#include <hip/hip_runtime.h>

#define B_  8
#define S_  2048
#define D_  1024

typedef unsigned short u16;
typedef __attribute__((ext_vector_type(16))) __bf16 v16bf;
typedef __attribute__((ext_vector_type(8)))  __bf16 v8bf;
typedef __attribute__((ext_vector_type(8)))  float  v8f;

union V16 { v16bf v; v8bf h[2]; };

__device__ inline u16 f2bf(float f) {
  unsigned u = __builtin_bit_cast(unsigned int, f);
  unsigned r = u + 0x7FFFu + ((u >> 16) & 1u);   // round-to-nearest-even
  return (u16)(r >> 16);
}

// ---------------- conversion / prep kernels ----------------

__global__ void cvt_bf16(const float* __restrict__ src, u16* __restrict__ dst, long n) {
  long i = (long)blockIdx.x * blockDim.x + threadIdx.x;
  long stride = (long)gridDim.x * blockDim.x;
  for (; i < n; i += stride) dst[i] = f2bf(src[i]);
}

__global__ void log_f32(const float* __restrict__ src, float* __restrict__ dst, long n) {
  long i = (long)blockIdx.x * blockDim.x + threadIdx.x;
  long stride = (long)gridDim.x * blockDim.x;
  for (; i < n; i += stride) dst[i] = __logf(src[i]);
}

// dst[b][d][s] = bf16(src[b][s][d])   (value transposed so P*V becomes an NT GEMM)
__global__ void cvt_transpose_v(const float* __restrict__ src, u16* __restrict__ dst) {
  __shared__ u16 tile[32][33];
  int b  = blockIdx.z;
  int d0 = blockIdx.x * 32;
  int s0 = blockIdx.y * 32;
  int tx = threadIdx.x;        // 0..31
  int ty = threadIdx.y;        // 0..7
  const float* sp = src + (size_t)b * S_ * D_;
  u16*         dp = dst + (size_t)b * D_ * S_;
#pragma unroll
  for (int i = 0; i < 4; ++i) {
    int sl = ty + i * 8;
    tile[sl][tx] = f2bf(sp[(size_t)(s0 + sl) * D_ + (d0 + tx)]);
  }
  __syncthreads();
#pragma unroll
  for (int i = 0; i < 4; ++i) {
    int dl = ty + i * 8;
    dp[(size_t)(d0 + dl) * S_ + (s0 + tx)] = tile[tx][dl];
  }
}

// ---------------- WMMA GEMM core: one wave -> 32(M) x 64(N) tile ----------------
// C[m][n] = sum_k A[m][k] * Brow[n][k], bf16 inputs, fp32 accumulate.
// 2-stage software pipeline with UNCONDITIONAL next-step prefetch: the k+32
// offset stays a compile-time immediate, so addressing is constant-folded and
// loadcnt never needs to drain inside the loop. The final iteration reads up
// to 64B past the logical end; all operands live in d_ws, which is allocated
// with guard padding to make those reads safe.

__device__ inline void gemm_core(const u16* __restrict__ A, const u16* __restrict__ Bm,
                                 int K, int lda, int ldb,
                                 int mBase, int nBase, v8f acc[2][4]) {
  int lane = threadIdx.x & 31;
  int mr   = lane & 15;
  int half = lane >> 4;
  const u16* arow0 = A  + (size_t)(mBase      + mr) * lda + half * 16;
  const u16* arow1 = A  + (size_t)(mBase + 16 + mr) * lda + half * 16;
  const u16* br0   = Bm + (size_t)(nBase +  0 + mr) * ldb + half * 16;
  const u16* br1   = Bm + (size_t)(nBase + 16 + mr) * ldb + half * 16;
  const u16* br2   = Bm + (size_t)(nBase + 32 + mr) * ldb + half * 16;
  const u16* br3   = Bm + (size_t)(nBase + 48 + mr) * ldb + half * 16;

  V16 a0, a1;
  v16bf b0, b1, b2, b3;
  // prologue: stage k0 = 0
  a0.h[0] = *(const v8bf*)(arow0);        a0.h[1] = *(const v8bf*)(arow0 + 8);
  a1.h[0] = *(const v8bf*)(arow1);        a1.h[1] = *(const v8bf*)(arow1 + 8);
  b0 = *(const v16bf*)(br0);
  b1 = *(const v16bf*)(br1);
  b2 = *(const v16bf*)(br2);
  b3 = *(const v16bf*)(br3);

  for (int k0 = 0; k0 < K; k0 += 32) {
    // issue next step's loads first (constant +32 offset; guarded over-read on last iter)
    V16 na0, na1;
    v16bf nb0, nb1, nb2, nb3;
    na0.h[0] = *(const v8bf*)(arow0 + k0 + 32); na0.h[1] = *(const v8bf*)(arow0 + k0 + 40);
    na1.h[0] = *(const v8bf*)(arow1 + k0 + 32); na1.h[1] = *(const v8bf*)(arow1 + k0 + 40);
    nb0 = *(const v16bf*)(br0 + k0 + 32);
    nb1 = *(const v16bf*)(br1 + k0 + 32);
    nb2 = *(const v16bf*)(br2 + k0 + 32);
    nb3 = *(const v16bf*)(br3 + k0 + 32);

    acc[0][0] = __builtin_amdgcn_wmma_f32_16x16x32_bf16(false, a0.v, false, b0, (short)0, acc[0][0], false, false);
    acc[0][1] = __builtin_amdgcn_wmma_f32_16x16x32_bf16(false, a0.v, false, b1, (short)0, acc[0][1], false, false);
    acc[0][2] = __builtin_amdgcn_wmma_f32_16x16x32_bf16(false, a0.v, false, b2, (short)0, acc[0][2], false, false);
    acc[0][3] = __builtin_amdgcn_wmma_f32_16x16x32_bf16(false, a0.v, false, b3, (short)0, acc[0][3], false, false);
    acc[1][0] = __builtin_amdgcn_wmma_f32_16x16x32_bf16(false, a1.v, false, b0, (short)0, acc[1][0], false, false);
    acc[1][1] = __builtin_amdgcn_wmma_f32_16x16x32_bf16(false, a1.v, false, b1, (short)0, acc[1][1], false, false);
    acc[1][2] = __builtin_amdgcn_wmma_f32_16x16x32_bf16(false, a1.v, false, b2, (short)0, acc[1][2], false, false);
    acc[1][3] = __builtin_amdgcn_wmma_f32_16x16x32_bf16(false, a1.v, false, b3, (short)0, acc[1][3], false, false);

    a0 = na0; a1 = na1;
    b0 = nb0; b1 = nb1; b2 = nb2; b3 = nb3;
  }
}

#define ACC_ZERO { { {0,0,0,0,0,0,0,0}, {0,0,0,0,0,0,0,0}, {0,0,0,0,0,0,0,0}, {0,0,0,0,0,0,0,0} }, \
                   { {0,0,0,0,0,0,0,0}, {0,0,0,0,0,0,0,0}, {0,0,0,0,0,0,0,0}, {0,0,0,0,0,0,0,0} } }

// Block = 256 thr = 8 waves; waves split M (256 rows), share one 64-wide N strip
// so the dominant B-fragment loads dedupe across waves in L0/L2.

__global__ void __launch_bounds__(256)
gemm_nt_bf16out(const u16* __restrict__ A, const u16* __restrict__ Bm,
                const float* __restrict__ bias, u16* __restrict__ C,
                int K, int lda, int ldb, int ldc) {
  int lane  = threadIdx.x & 31;
  int wave  = threadIdx.x >> 5;
  int mBase = blockIdx.x * 256 + wave * 32;
  int nBase = blockIdx.y * 64;
  v8f acc[2][4] = ACC_ZERO;
  gemm_core(A, Bm, K, lda, ldb, mBase, nBase, acc);
  int half = lane >> 4;
#pragma unroll
  for (int s = 0; s < 2; ++s) {
#pragma unroll
    for (int j = 0; j < 4; ++j) {
      int n = nBase + j * 16 + (lane & 15);
      float bv = bias ? bias[n] : 0.0f;
#pragma unroll
      for (int r = 0; r < 8; ++r) {
        int m = mBase + s * 16 + half * 8 + r;
        C[(size_t)m * ldc + n] = f2bf(acc[s][j][r] + bv);
      }
    }
  }
}

__global__ void __launch_bounds__(256)
gemm_nt_f32out(const u16* __restrict__ A, const u16* __restrict__ Bm,
               const float* __restrict__ bias, float* __restrict__ C,
               int K, int lda, int ldb, int ldc) {
  int lane  = threadIdx.x & 31;
  int wave  = threadIdx.x >> 5;
  int mBase = blockIdx.x * 256 + wave * 32;
  int nBase = blockIdx.y * 64;
  v8f acc[2][4] = ACC_ZERO;
  gemm_core(A, Bm, K, lda, ldb, mBase, nBase, acc);
  int half = lane >> 4;
#pragma unroll
  for (int s = 0; s < 2; ++s) {
#pragma unroll
    for (int j = 0; j < 4; ++j) {
      int n = nBase + j * 16 + (lane & 15);
      float bv = bias[n];
#pragma unroll
      for (int r = 0; r < 8; ++r) {
        int m = mBase + s * 16 + half * 8 + r;
        C[(size_t)m * ldc + n] = acc[s][j][r] + bv;
      }
    }
  }
}

// scores[m][n] = Q[m]·Kp[n] + log(cell_mask[m][n]) + log(seq_mask[n])   (one batch)
__global__ void __launch_bounds__(256)
gemm_scores(const u16* __restrict__ Q, const u16* __restrict__ Kp,
            const float* __restrict__ cm, const float* __restrict__ lsmB,
            float* __restrict__ Sout) {
  int lane  = threadIdx.x & 31;
  int wave  = threadIdx.x >> 5;
  int mBase = blockIdx.x * 256 + wave * 32;
  int nBase = blockIdx.y * 64;
  v8f acc[2][4] = ACC_ZERO;
  gemm_core(Q, Kp, D_, D_, D_, mBase, nBase, acc);
  int half = lane >> 4;
#pragma unroll
  for (int s = 0; s < 2; ++s) {
#pragma unroll
    for (int j = 0; j < 4; ++j) {
      int n = nBase + j * 16 + (lane & 15);
      float ls = lsmB[n];
#pragma unroll
      for (int r = 0; r < 8; ++r) {
        int m = mBase + s * 16 + half * 8 + r;
        size_t idx = (size_t)m * S_ + n;
        Sout[idx] = acc[s][j][r] + __logf(cm[idx]) + ls;
      }
    }
  }
}

// Row softmax over 2048 cols; writes probs in place as bf16 (row stride stays 2048 floats).
__global__ void __launch_bounds__(256)
softmax_rows(float* __restrict__ scores) {
  __shared__ float red[256];
  int row = blockIdx.x;
  int t   = threadIdx.x;
  float* srow = scores + (size_t)row * S_;
  float v[8];
  float lmax = -3.402823466e38f;
#pragma unroll
  for (int i = 0; i < 8; ++i) { v[i] = srow[t + i * 256]; lmax = fmaxf(lmax, v[i]); }
  red[t] = lmax; __syncthreads();
  for (int s = 128; s > 0; s >>= 1) { if (t < s) red[t] = fmaxf(red[t], red[t + s]); __syncthreads(); }
  float m = red[0]; __syncthreads();
  float lsum = 0.0f;
#pragma unroll
  for (int i = 0; i < 8; ++i) { v[i] = __expf(v[i] - m); lsum += v[i]; }
  red[t] = lsum; __syncthreads();
  for (int s = 128; s > 0; s >>= 1) { if (t < s) red[t] += red[t + s]; __syncthreads(); }
  float inv = 1.0f / red[0];
  __syncthreads();                 // all reads of the fp32 row are complete before bf16 overwrite
  u16* prow = (u16*)srow;
#pragma unroll
  for (int i = 0; i < 8; ++i) prow[t + i * 256] = f2bf(v[i] * inv);
}

// ---------------- host ----------------

extern "C" void kernel_launch(void* const* d_in, const int* in_sizes, int n_in,
                              void* d_out, int out_size, void* d_ws, size_t ws_size,
                              hipStream_t stream) {
  const float* q   = (const float*)d_in[0];
  const float* key = (const float*)d_in[1];
  const float* val = (const float*)d_in[2];
  const float* cm  = (const float*)d_in[3];
  const float* sm  = (const float*)d_in[4];
  const float* Wk  = (const float*)d_in[5];
  const float* bk  = (const float*)d_in[6];
  const float* Wo  = (const float*)d_in[7];
  const float* bo  = (const float*)d_in[8];

  char* ws = (char*)d_ws;
  size_t off = 0;
  // every allocation gets a 256B guard so the pipelined GEMM's trailing
  // 64B over-read on the final K-step stays inside the workspace
  auto alloc = [&](size_t bytes) -> void* {
    void* p = ws + off;
    off += ((bytes + 255) & ~(size_t)255) + 256;
    return p;
  };
  const size_t BSD = (size_t)B_ * S_ * D_;
  u16*   Qb  = (u16*)alloc(BSD * 2);
  u16*   Kb  = (u16*)alloc(BSD * 2);
  u16*   Vt  = (u16*)alloc(BSD * 2);          // transposed: [b][d][s]
  u16*   KPb = (u16*)alloc(BSD * 2);          // k_proj bf16
  u16*   Xb  = (u16*)alloc(BSD * 2);          // attention output bf16
  u16*   Wkb = (u16*)alloc((size_t)D_ * D_ * 2);
  u16*   Wob = (u16*)alloc((size_t)D_ * D_ * 2);
  float* lsm = (float*)alloc((size_t)B_ * S_ * 4);
  float* Sc  = (float*)alloc((size_t)S_ * S_ * 4);   // per-batch score/prob buffer (reused)

  // 1) precision conversion + V transpose + log(seq_mask)
  cvt_bf16<<<4096, 256, 0, stream>>>(q,   Qb, (long)BSD);
  cvt_bf16<<<4096, 256, 0, stream>>>(key, Kb, (long)BSD);
  cvt_bf16<<<1024, 256, 0, stream>>>(Wk,  Wkb, (long)D_ * D_);
  cvt_bf16<<<1024, 256, 0, stream>>>(Wo,  Wob, (long)D_ * D_);
  cvt_transpose_v<<<dim3(D_ / 32, S_ / 32, B_), dim3(32, 8), 0, stream>>>(val, Vt);
  log_f32<<<64, 256, 0, stream>>>(sm, lsm, (long)B_ * S_);

  // 2) k_proj = key @ W_k^T + b_k   (batch folded into M)
  gemm_nt_bf16out<<<dim3(B_ * S_ / 256, D_ / 64), 256, 0, stream>>>(
      Kb, Wkb, bk, KPb, D_, D_, D_, D_);

  // 3) per-batch: scores -> softmax -> P*V
  for (int b = 0; b < B_; ++b) {
    const size_t ob = (size_t)b * S_ * D_;
    gemm_scores<<<dim3(S_ / 256, S_ / 64), 256, 0, stream>>>(
        Qb + ob, KPb + ob, cm + (size_t)b * S_ * S_, lsm + (size_t)b * S_, Sc);
    softmax_rows<<<S_, 256, 0, stream>>>(Sc);
    // P (bf16, row stride 2*S_ u16 elems, aliased on Sc) times V^T rows
    gemm_nt_bf16out<<<dim3(S_ / 256, D_ / 64), 256, 0, stream>>>(
        (const u16*)Sc, Vt + ob, (const float*)nullptr, Xb + ob,
        S_, 2 * S_, S_, D_);
  }

  // 4) out = x @ W_o^T + b_o   (batch folded into M), fp32 output
  gemm_nt_f32out<<<dim3(B_ * S_ / 256, D_ / 64), 256, 0, stream>>>(
      Xb, Wob, bo, (float*)d_out, D_, D_, D_, D_);
}
